// BiLSTM_4930622456140
// MI455X (gfx1250) — compile-verified, matrix operand order checked by reading, not run on hardware
//
#include <hip/hip_runtime.h>

// ---------------------------------------------------------------------------
// BiLSTM (2-layer bidirectional, Keras gates i,f,g,o, residual on layer 1,
// merge='ave') for MI455X / gfx1250.  All matmuls: V_WMMA_F32_16X16X32_BF16.
//
//  Phase 1: xz = bf16(x) @ bf16(Wx) + b   -- WMMA GEMM over transposed rows
//           m' = t*256 + b, OUTPUT WRITTEN IN PACKED C-FRAGMENT ORDER
//           [m'tile][ntile][lane][8f] so the recurrence reads it with one
//           v8f load per gate.
//  Phase 2: per-step z = xz_t + h @ Wh; gates; c,h   (serial path:
//           - 256 KB of packed Wh pinned in LDS (of 320 KB/WGP)
//           - h double-buffered in LDS -> ONE barrier per timestep)
//  Phase 3: out = 0.5*(h_fw1 + h_fw0 + h_bw1 + h_bw0)
//
// Async staging: GLOBAL_LOAD_ASYNC_TO_LDS_B128 + s_wait_asynccnt (builtin is
// present on this toolchain with generic v4i* params; guarded fallback kept).
// Workspace: ~1.20 GiB.
// ---------------------------------------------------------------------------

#define DEV __device__ __forceinline__

typedef __attribute__((ext_vector_type(16))) __bf16        v16bf;
typedef __attribute__((ext_vector_type(8)))  float         v8f;
typedef __attribute__((ext_vector_type(4)))  unsigned int  u32x4;

union Frag16 { v16bf v; u32x4 q[2]; };

// ---- async copy to LDS (gfx1250) with compile-safe fallback ---------------
#if defined(__has_builtin)
#  if __has_builtin(__builtin_amdgcn_global_load_async_to_lds_b128)
#    define HAVE_ASYNC_LDS 1
#  endif
#endif
#ifndef HAVE_ASYNC_LDS
#  define HAVE_ASYNC_LDS 0
#endif

#if HAVE_ASYNC_LDS
// Probe-observed parameter type: generic pointer to 'int __vector(4)'.
typedef __attribute__((__vector_size__(4 * sizeof(int)))) int v4i_b;
DEV void cp16_async(const void* g, void* l) {   // 16B global -> LDS, ASYNCcnt
  __builtin_amdgcn_global_load_async_to_lds_b128(
      (v4i_b*)(uintptr_t)g, (v4i_b*)(uintptr_t)l, 0, 0);
}
DEV void async_wait0() {
#  if __has_builtin(__builtin_amdgcn_s_wait_asynccnt)
  __builtin_amdgcn_s_wait_asynccnt(0);
#  else
  asm volatile("s_wait_asynccnt 0x0" ::: "memory");
#  endif
}
#else
DEV void cp16_async(const void* g, void* l) { *(u32x4*)l = *(const u32x4*)g; }
DEV void async_wait0() {}
#endif

DEV unsigned short f2bf(float f) {           // RNE fp32 -> bf16
  unsigned int u = __float_as_uint(f);
  u += 0x7FFFu + ((u >> 16) & 1u);
  return (unsigned short)(u >> 16);
}

DEV float sigmoidf_(float x) { return 1.0f / (1.0f + __expf(-x)); }
DEV float tanhf_(float x) {
  float a = fabsf(x);
  float e = __expf(-2.0f * a);
  float t = (1.0f - e) / (1.0f + e);
  return copysignf(t, x);
}

// Packed B fragment (contiguous 32B per lane) from global.
DEV v16bf load_bfrag_global(const unsigned short* __restrict__ base, int tile, int lane) {
  return *(const v16bf*)(base + (size_t)tile * 512 + lane * 16);
}

// Packed B fragment from LDS cache (contiguous): q[1] at +8 elements.
DEV v16bf load_bfrag_lds(const unsigned short* p) {
  Frag16 fr;
  fr.q[0] = *(const u32x4*)(p);
  fr.q[1] = *(const u32x4*)(p + 8);
  return fr.v;
}

// A fragment from row-major LDS tile (16 rows x 32 k), ISA 16-bit A layout:
// lanes 0-15 (row m=L) hold K {0..7,16..23}; lanes 16-31 hold {8..15,24..31}.
DEV v16bf load_afrag_lds(const unsigned short* p_kslice, int pitch, int lane) {
  const int m   = lane & 15;
  const int seg = (lane >> 4) << 3;          // 0 or 8 elements
  const unsigned short* p = p_kslice + m * pitch + seg;
  Frag16 fr;
  fr.q[0] = *(const u32x4*)(p);              // K seg .. seg+7
  fr.q[1] = *(const u32x4*)(p + 16);         // K seg+16 .. seg+23 (gap of 16)
  return fr.v;
}

// ---------------------------------------------------------------------------
// Pack fp32 weight [K=256, N=1024] row-major into WMMA fragment order:
// [ntile(64)][ktile(8)][lane(32) * 16 bf16].
// ---------------------------------------------------------------------------
__global__ void pack_weight(const float* __restrict__ W, unsigned short* __restrict__ P) {
  int p    = blockIdx.x * blockDim.x + threadIdx.x;     // 262144 total
  int tile = p >> 9;
  int r    = p & 511;
  int L    = r >> 4, e = r & 15;
  int nt   = tile >> 3, kt = tile & 7;
  int kk   = (e < 8 ? e : e + 8) + ((L >= 16) ? 8 : 0);
  P[p] = f2bf(W[(kt * 32 + kk) * 1024 + nt * 16 + (L & 15)]);
}

__global__ void f32_to_bf16(const float* __restrict__ src, unsigned short* __restrict__ dst,
                            size_t n) {
  size_t i = (size_t)blockIdx.x * blockDim.x + threadIdx.x;
  if (i < n) dst[i] = f2bf(src[i]);
}

__global__ void merge_avg(const float* __restrict__ a, const float* __restrict__ b,
                          const float* __restrict__ c, const float* __restrict__ d,
                          float* __restrict__ out, size_t n) {
  size_t i = (size_t)blockIdx.x * blockDim.x + threadIdx.x;
  if (i < n) out[i] = 0.5f * (a[i] + b[i] + c[i] + d[i]);
}

// ---------------------------------------------------------------------------
// xz GEMM over logically transposed rows m' = t*256 + b  (b = m'&255,
// t = m'>>8; A source is [b][t][256] bf16).  Output in packed C-fragment
// order: Cp[(m'tile*64 + ntile)*256 + lane*8 + j].
// 128x128 tile/WG, 8 waves, A double-buffered in LDS, async staged.
// ---------------------------------------------------------------------------
__global__ __launch_bounds__(256) void wmma_gemm_xz(
    const unsigned short* __restrict__ A,   // [b][t][256] bf16
    const unsigned short* __restrict__ Wp,  // packed Wx fragments
    const float* __restrict__ bias,         // [1024]
    float* __restrict__ Cp) {               // packed xz fragments
  __shared__ unsigned short atile[2][128 * 48];
  const int tid  = threadIdx.x;
  const int lane = tid & 31;
  const int w    = tid >> 5;
  const int wm   = w & 3;                 // 32-row group
  const int wn   = w >> 2;                // 64-col group
  const int m0   = blockIdx.y * 128;      // m' base
  const int nblk = blockIdx.x;            // 0..7

  v8f acc[2][4];
#pragma unroll
  for (int mt = 0; mt < 2; ++mt)
#pragma unroll
    for (int nt = 0; nt < 4; ++nt)
#pragma unroll
      for (int j = 0; j < 8; ++j) acc[mt][nt][j] = 0.0f;

  const int srow = tid >> 1, shalf = tid & 1;
  const int rowp = m0 + srow;                              // m'
  const size_t arow = ((size_t)(rowp & 255) * 512 + (rowp >> 8)) * 256;
  const unsigned short* asrc = A + arow + shalf * 16;

  {   // prologue: stage K-slice 0 (async -> LDS)
    unsigned short* dst = atile[0] + srow * 48 + shalf * 16;
    cp16_async(asrc, dst);
    cp16_async(asrc + 8, dst + 8);
  }

  for (int kt = 0; kt < 8; ++kt) {
    async_wait0();
    __syncthreads();                       // slice kt staged & prev reads done
    if (kt < 7) {                          // stage kt+1 into other buffer
      const unsigned short* src = asrc + (kt + 1) * 32;
      unsigned short* dst = atile[(kt + 1) & 1] + srow * 48 + shalf * 16;
      cp16_async(src, dst);
      cp16_async(src + 8, dst + 8);
    }
    v16bf bfr[4];
#pragma unroll
    for (int nt = 0; nt < 4; ++nt) {
      int ntg = nblk * 8 + wn * 4 + nt;
      bfr[nt] = load_bfrag_global(Wp, ntg * 8 + kt, lane);
    }
#pragma unroll
    for (int mt = 0; mt < 2; ++mt) {
      v16bf af = load_afrag_lds(atile[kt & 1] + (wm * 32 + mt * 16) * 48, 48, lane);
#pragma unroll
      for (int nt = 0; nt < 4; ++nt)
        acc[mt][nt] = __builtin_amdgcn_wmma_f32_16x16x32_bf16(
            false, af, false, bfr[nt], (short)0, acc[mt][nt], false, false);
    }
  }
  // epilogue: packed fragment store, one v8f (32B) per lane per tile.
#pragma unroll
  for (int mt = 0; mt < 2; ++mt) {
    int mtile = blockIdx.y * 8 + wm * 2 + mt;              // m'/16
#pragma unroll
    for (int nt = 0; nt < 4; ++nt) {
      int ntg  = nblk * 8 + wn * 4 + nt;
      float bv = bias[ntg * 16 + (lane & 15)];
      v8f r;
#pragma unroll
      for (int j = 0; j < 8; ++j) r[j] = acc[mt][nt][j] + bv;
      *(v8f*)(Cp + ((size_t)mtile * 64 + ntg) * 256 + lane * 8) = r;
    }
  }
}

// ---------------------------------------------------------------------------
// LSTM recurrence, 16 batch rows / WG, 16 waves.  Wave w owns units
// u in [16w,16w+16) and gate-aligned n-tiles {w,16+w,32+w,48+w}.
// xz arrives pre-packed: one v8f load per gate per step.
// Dynamic LDS (279,040 B < 320 KB/WGP): 256 KB Wh pin + 2 h buffers.
// ---------------------------------------------------------------------------
#define REC_WHC_ELEMS   (64 * 4 * 512)       // 131072 bf16 = 256 KiB
#define REC_HBUF_ELEMS  (16 * 264)           // 4224  bf16
#define REC_LDS_BYTES   ((REC_WHC_ELEMS + 2 * REC_HBUF_ELEMS) * 2)

__global__ __launch_bounds__(512) void wmma_lstm_rec(
    const float* __restrict__ xzp,           // packed fragments
    const unsigned short* __restrict__ Whp,  // packed Wh fragments
    float* __restrict__ h_out,               // [B,T,256] fp32
    unsigned short* __restrict__ hbf_out,    // [B,T,256] bf16
    int reverse) {
  extern __shared__ unsigned short smem[];
  unsigned short* whc   = smem;
  unsigned short* hbuf0 = smem + REC_WHC_ELEMS;
  unsigned short* hbuf1 = hbuf0 + REC_HBUF_ELEMS;

  const int tid  = threadIdx.x;
  const int lane = tid & 31;
  const int w    = tid >> 5;                 // 0..15 unit group
  const int bt   = blockIdx.x;               // batch tile
  const int b0   = bt * 16;
  const int rlo  = (lane >> 4) << 3;
  const int u    = w * 16 + (lane & 15);

  // Pin Wh k-slices 0..3 in LDS (async copy in packed fragment order).
  for (int i = tid; i < REC_WHC_ELEMS / 8; i += 512) {
    int e    = i << 3;
    int tile = e >> 9;                       // ntg*4 + kt
    int ntg  = tile >> 2, kt = tile & 3;
    cp16_async(Whp + (size_t)((ntg * 8 + kt) << 9) + (e & 511), whc + e);
  }
  for (int i = tid; i < 2 * REC_HBUF_ELEMS; i += 512) hbuf0[i] = 0;  // h0 = 0

  float c[8];
#pragma unroll
  for (int j = 0; j < 8; ++j) c[j] = 0.0f;                           // c0 = 0

  async_wait0();
  __syncthreads();

  for (int s = 0; s < 512; ++s) {
    const int t = reverse ? (511 - s) : s;
    unsigned short* hcur = (s & 1) ? hbuf1 : hbuf0;
    unsigned short* hnxt = (s & 1) ? hbuf0 : hbuf1;

    // C accumulators straight from packed xz: one 32B load per gate.
    const size_t tb = ((size_t)t * 16 + bt) * 64;
    v8f acc[4];
#pragma unroll
    for (int g = 0; g < 4; ++g)
      acc[g] = *(const v8f*)(xzp + (tb + (g * 16 + w)) * 256 + lane * 8);
    if (s + 1 < 512) {   // gfx1250 global_prefetch of next step's xz tile
      const int tn = reverse ? (511 - (s + 1)) : (s + 1);
      __builtin_prefetch(xzp + (((size_t)tn * 16 + bt) * 64 + w) * 256 + lane * 8, 0, 0);
    }

    // z += h @ Wh : k-slices 0..3 B-frags from LDS, 4..7 from L2.
#pragma unroll
    for (int kt = 0; kt < 8; ++kt) {
      v16bf af = load_afrag_lds(hcur + kt * 32, 264, lane);
#pragma unroll
      for (int g = 0; g < 4; ++g) {
        int ntg = g * 16 + w;
        v16bf bf_ = (kt < 4)
            ? load_bfrag_lds(whc + ((ntg * 4 + kt) << 9) + lane * 16)
            : load_bfrag_global(Whp, ntg * 8 + kt, lane);
        acc[g] = __builtin_amdgcn_wmma_f32_16x16x32_bf16(
            false, af, false, bf_, (short)0, acc[g], false, false);
      }
    }

    // gates + state update; write h into the OTHER buffer (no read conflict)
#pragma unroll
    for (int j = 0; j < 8; ++j) {
      float ig = sigmoidf_(acc[0][j]);
      float fg = sigmoidf_(acc[1][j]);
      float gg = tanhf_(acc[2][j]);
      float og = sigmoidf_(acc[3][j]);
      float cj = fg * c[j] + ig * gg;
      c[j] = cj;
      float hj = og * tanhf_(cj);
      size_t oidx = ((size_t)(b0 + rlo + j) * 512 + t) * 256 + u;
      h_out[oidx]   = hj;
      unsigned short hb = f2bf(hj);
      hbf_out[oidx] = hb;
      hnxt[(rlo + j) * 264 + u] = hb;
    }
    __syncthreads();   // single barrier per timestep
  }
}

// ---------------------------------------------------------------------------
extern "C" void kernel_launch(void* const* d_in, const int* in_sizes, int n_in,
                              void* d_out, int out_size, void* d_ws, size_t ws_size,
                              hipStream_t stream) {
  (void)in_sizes; (void)n_in; (void)out_size; (void)ws_size;

  const size_t WMAT = 256 * 1024;                    // elems per weight matrix
  const size_t NX   = (size_t)256 * 512 * 256;       // B*T*U

  const float* x = (const float*)d_in[0];
  const float* Wxs[4] = {(const float*)d_in[1], (const float*)d_in[4],
                         (const float*)d_in[7], (const float*)d_in[10]};
  const float* Whs[4] = {(const float*)d_in[2], (const float*)d_in[5],
                         (const float*)d_in[8], (const float*)d_in[11]};
  const float* bs[4]  = {(const float*)d_in[3], (const float*)d_in[6],
                         (const float*)d_in[9], (const float*)d_in[12]};

  unsigned short* u16ws = (unsigned short*)d_ws;
  unsigned short* wpack = u16ws;               // 8 * WMAT bf16   (4 MiB)
  unsigned short* xbf   = wpack + 8 * WMAT;    // NX bf16
  unsigned short* hbff  = xbf + NX;            // NX bf16
  unsigned short* hbfb  = hbff + NX;           // NX bf16
  float* hf0 = (float*)(hbfb + NX);
  float* hb0 = hf0 + NX;
  float* hf1 = hb0 + NX;
  float* hb1 = hf1 + NX;
  float* xzp = hb1 + NX;                       // packed xz (512 MiB, reused)

  (void)hipFuncSetAttribute(reinterpret_cast<const void*>(&wmma_lstm_rec),
                            hipFuncAttributeMaxDynamicSharedMemorySize,
                            REC_LDS_BYTES);

  for (int d = 0; d < 4; ++d) {
    pack_weight<<<1024, 256, 0, stream>>>(Wxs[d], wpack + (size_t)(2 * d) * WMAT);
    pack_weight<<<1024, 256, 0, stream>>>(Whs[d], wpack + (size_t)(2 * d + 1) * WMAT);
  }
  f32_to_bf16<<<(unsigned)(NX / 256), 256, 0, stream>>>(x, xbf, NX);

  dim3 ggrid(8, 1024);   // N/128 x M/128

  // layer 0 fw / bw (stream-serialized -> single xzp buffer is safe)
  wmma_gemm_xz<<<ggrid, 256, 0, stream>>>(xbf, wpack + 0 * WMAT, bs[0], xzp);
  wmma_lstm_rec<<<16, 512, REC_LDS_BYTES, stream>>>(xzp, wpack + 1 * WMAT, hf0, hbff, 0);
  wmma_gemm_xz<<<ggrid, 256, 0, stream>>>(xbf, wpack + 2 * WMAT, bs[1], xzp);
  wmma_lstm_rec<<<16, 512, REC_LDS_BYTES, stream>>>(xzp, wpack + 3 * WMAT, hb0, hbfb, 1);

  // layer 1 (inputs = layer-0 outputs; residual folded into merge)
  wmma_gemm_xz<<<ggrid, 256, 0, stream>>>(hbff, wpack + 4 * WMAT, bs[2], xzp);
  wmma_lstm_rec<<<16, 512, REC_LDS_BYTES, stream>>>(xzp, wpack + 5 * WMAT, hf1, hbff, 0);
  wmma_gemm_xz<<<ggrid, 256, 0, stream>>>(hbfb, wpack + 6 * WMAT, bs[3], xzp);
  wmma_lstm_rec<<<16, 512, REC_LDS_BYTES, stream>>>(xzp, wpack + 7 * WMAT, hb1, hbfb, 1);

  merge_avg<<<(unsigned)(NX / 256), 256, 0, stream>>>(hf1, hf0, hb1, hb0,
                                                      (float*)d_out, NX);
}
